// RelAttention_71725953843606
// MI455X (gfx1250) — matrix operand order repaired and checked by reading, not compiled, not run on hardware
//
#include <hip/hip_runtime.h>
#include <math.h>
#include <stdint.h>

#define B_SZ   16
#define N_SEQ  2048
#define DK     256
#define BM     64
#define BN     64
#define NTHR   256
#define NTILE  (N_SEQ / BN)   // 32 key tiles per batch

typedef __bf16 v16bf __attribute__((ext_vector_type(16)));
typedef float  v8f   __attribute__((ext_vector_type(8)));

union Frag { v16bf v; uint4 q[2]; };

__device__ __forceinline__ unsigned short f2bf(float f) {
  unsigned int u = __float_as_uint(f);
  u += 0x7FFFu + ((u >> 16) & 1u);      // round to nearest even
  return (unsigned short)(u >> 16);
}

// ---- CDNA5 async global->LDS copy (16B per lane, ASYNCcnt tracked) ----
__device__ __forceinline__ void async_copy16(unsigned lds_addr, unsigned voff,
                                             unsigned long long sbase) {
  asm volatile("global_load_async_to_lds_b128 %0, %1, %2"
               :: "v"(lds_addr), "v"(voff), "s"(sbase) : "memory");
}
__device__ __forceinline__ void wait_async0() {
#if __has_builtin(__builtin_amdgcn_s_wait_asynccnt)
  __builtin_amdgcn_s_wait_asynccnt(0);
#else
  asm volatile("s_wait_asynccnt 0x0" ::: "memory");
#endif
}

// ================= prepass: f32 -> bf16 (row-major), grid-stride =================
__global__ __launch_bounds__(NTHR)
void cvt_bf16_kernel(const float4* __restrict__ src, uint2* __restrict__ dst, int n4) {
  int i = blockIdx.x * blockDim.x + threadIdx.x;
  int stride = gridDim.x * blockDim.x;
  for (; i < n4; i += stride) {
    float4 x = src[i];
    uint2 u;
    u.x = (unsigned)f2bf(x.x) | ((unsigned)f2bf(x.y) << 16);
    u.y = (unsigned)f2bf(x.z) | ((unsigned)f2bf(x.w) << 16);
    dst[i] = u;
  }
}

// ===== prepass: V -> bf16, transposed per 64-row tile: Vt[tile][d*64 + r] =====
__global__ __launch_bounds__(NTHR)
void cvt_vt_kernel(const float* __restrict__ V, unsigned short* __restrict__ Vt) {
  __shared__ __align__(16) unsigned short sVt[DK * BN];
  const int b = blockIdx.y, jt = blockIdx.x, tid = threadIdx.x;
  const float4* vs = (const float4*)(V + ((size_t)b * N_SEQ + (size_t)jt * BN) * DK);
#pragma unroll 4
  for (int k = 0; k < 16; ++k) {
    int idx = tid + k * NTHR;
    int row = idx >> 6;
    int c   = (idx & 63) << 2;
    float4 y = vs[idx];
    sVt[(c + 0) * BN + row] = f2bf(y.x);
    sVt[(c + 1) * BN + row] = f2bf(y.y);
    sVt[(c + 2) * BN + row] = f2bf(y.z);
    sVt[(c + 3) * BN + row] = f2bf(y.w);
  }
  __syncthreads();
  uint4* out = (uint4*)(Vt + (size_t)(b * NTILE + jt) * (DK * BN));
  const uint4* s = (const uint4*)sVt;
#pragma unroll
  for (int k = 0; k < 8; ++k) out[tid + k * NTHR] = s[tid + k * NTHR];
}

// ========================== fused flash attention ==========================
template <bool PRE>
__global__ __launch_bounds__(NTHR)
void relattn_fa_kernel(const float* __restrict__ Q,
                       const float* __restrict__ K,
                       const float* __restrict__ V,
                       const float* __restrict__ PE,
                       float* __restrict__ O,
                       const unsigned short* __restrict__ Qbf,
                       const unsigned short* __restrict__ Kbf,
                       const unsigned short* __restrict__ Vtb)
{
  __shared__ __align__(16) unsigned short sQ [BM * DK];   // 32 KB bf16 row-major
  __shared__ __align__(16) unsigned short sK [BN * DK];   // 32 KB bf16 row-major
  __shared__ __align__(16) unsigned short sVt[DK * BN];   // 32 KB bf16 transposed
  __shared__ __align__(16) unsigned short sP [BM * BN];   //  8 KB bf16 probs
  __shared__ float sMax[BM * 2];
  __shared__ float sSum[BM * 2];

  const int b    = blockIdx.y;
  const int i0   = blockIdx.x * BM;       // first query row of this block
  const int tid  = threadIdx.x;
  const int w    = tid >> 5;              // wave 0..7
  const int lane = tid & 31;
  const int rw   = w >> 1;                // 16-row block 0..3
  const int ch   = w & 1;                 // column / d half
  const int half = lane >> 4;             // C-layout M split
  const int l16  = lane & 15;

  const float* Qb  = Q  + ((size_t)b * N_SEQ + i0) * DK;
  const float* Kb  = K  + (size_t)b * N_SEQ * DK;
  const float* Vb  = V  + (size_t)b * N_SEQ * DK;
  const float* PEb = PE + ((size_t)b * N_SEQ + i0) * N_SEQ;
  float*       Ob  = O  + ((size_t)b * N_SEQ + i0) * DK;

  // ---- stage Q tile into LDS ----
  if (PRE) {
    const unsigned ldsQ = (unsigned)(uintptr_t)&sQ[0];
    const unsigned long long baseQ =
        (unsigned long long)(uintptr_t)(Qbf + ((size_t)b * N_SEQ + i0) * DK);
#pragma unroll
    for (int k = 0; k < 8; ++k) {
      unsigned off = (unsigned)(tid + k * NTHR) * 16u;   // 32 KB total
      async_copy16(ldsQ + off, off, baseQ);
    }
    // completion folded into first iteration's wait_async0()
  } else {
    const float4* src = (const float4*)Qb;
#pragma unroll 4
    for (int k = 0; k < 16; ++k) {
      int idx = tid + k * NTHR;
      int row = idx >> 6;
      int c   = (idx & 63) << 2;
      float4 x = src[idx];
      uint2 u;
      u.x = (unsigned)f2bf(x.x) | ((unsigned)f2bf(x.y) << 16);
      u.y = (unsigned)f2bf(x.z) | ((unsigned)f2bf(x.w) << 16);
      *(uint2*)&sQ[row * DK + c] = u;
    }
  }

  v8f   acc[8];                // O accumulators: 8 d-tiles of 16x16 f32
  float m_run[8], l_run[8];    // stats, row = rw*16 + r + 8*half
#pragma unroll
  for (int t = 0; t < 8; ++t) acc[t] = {};
#pragma unroll
  for (int r = 0; r < 8; ++r) { m_run[r] = -3.0e38f; l_run[r] = 0.f; }

  const int ntiles = i0 / BN + 1;         // causal: only key tiles j0 <= i0

  for (int jt = 0; jt < ntiles; ++jt) {
    const int j0 = jt * BN;
    __syncthreads();                      // K/V/P LDS safe to overwrite

    // ---- stage K tile (row-major bf16) and V tile (transposed bf16) ----
    if (PRE) {
      if (jt + 1 < ntiles) {              // prefetch next bf16 tiles (32 KB each)
        const char* nk = (const char*)(Kbf + ((size_t)b * N_SEQ + j0 + BN) * DK);
        const char* nv = (const char*)(Vtb + (size_t)(b * NTILE + jt + 1) * (DK * BN));
        __builtin_prefetch(nk + tid * 128, 0, 0);
        __builtin_prefetch(nv + tid * 128, 0, 0);
      }
      const unsigned ldsK = (unsigned)(uintptr_t)&sK[0];
      const unsigned ldsV = (unsigned)(uintptr_t)&sVt[0];
      const unsigned long long baseK =
          (unsigned long long)(uintptr_t)(Kbf + ((size_t)b * N_SEQ + j0) * DK);
      const unsigned long long baseV =
          (unsigned long long)(uintptr_t)(Vtb + (size_t)(b * NTILE + jt) * (DK * BN));
#pragma unroll
      for (int k = 0; k < 8; ++k) {
        unsigned off = (unsigned)(tid + k * NTHR) * 16u;
        async_copy16(ldsK + off, off, baseK);
        async_copy16(ldsV + off, off, baseV);
      }
      wait_async0();
    } else {
      if (jt + 1 < ntiles) {
        const float* nk = Kb + (size_t)(j0 + BN) * DK;
        const float* nv = Vb + (size_t)(j0 + BN) * DK;
        __builtin_prefetch(nk + tid * 64, 0, 0);
        __builtin_prefetch(nv + tid * 64, 0, 0);
      }
      const float4* ks = (const float4*)(Kb + (size_t)j0 * DK);
      const float4* vs = (const float4*)(Vb + (size_t)j0 * DK);
#pragma unroll 4
      for (int k = 0; k < 16; ++k) {
        int idx = tid + k * NTHR;
        int row = idx >> 6;
        int c   = (idx & 63) << 2;
        float4 x = ks[idx];
        uint2 u;
        u.x = (unsigned)f2bf(x.x) | ((unsigned)f2bf(x.y) << 16);
        u.y = (unsigned)f2bf(x.z) | ((unsigned)f2bf(x.w) << 16);
        *(uint2*)&sK[row * DK + c] = u;
        float4 y = vs[idx];
        sVt[(c + 0) * BN + row] = f2bf(y.x);
        sVt[(c + 1) * BN + row] = f2bf(y.y);
        sVt[(c + 2) * BN + row] = f2bf(y.z);
        sVt[(c + 3) * BN + row] = f2bf(y.w);
      }
    }
    __syncthreads();

    // ---- S = Q*K^T : this wave's two 16x16 col-tiles, K-dim 256 = 8 steps ----
    v8f sacc[2];
#pragma unroll
    for (int t = 0; t < 2; ++t) {
      const int n0 = ch * 32 + t * 16;
      v8f c = {};
#pragma unroll
      for (int s = 0; s < 8; ++s) {
        Frag a, bb;
        const unsigned short* qrow = &sQ[(rw * 16 + l16) * DK + s * 32];
        a.q[0] = *(const uint4*)(qrow + half * 8);
        a.q[1] = *(const uint4*)(qrow + 16 + half * 8);
        const unsigned short* krow = &sK[(n0 + l16) * DK + s * 32];
        bb.q[0] = *(const uint4*)(krow + half * 8);
        bb.q[1] = *(const uint4*)(krow + 16 + half * 8);
        c = __builtin_amdgcn_wmma_f32_16x16x32_bf16(false, a.v, false, bb.v,
                                                    (short)0, c, false, false);
      }
      sacc[t] = c;
    }

    // ---- bias + causal mask + 1/sqrt(256) scale, in place ----
#pragma unroll
    for (int t = 0; t < 2; ++t) {
#pragma unroll
      for (int r = 0; r < 8; ++r) {
        int m = rw * 16 + r + 8 * half;             // row within q-block
        int n = j0 + ch * 32 + t * 16 + l16;        // absolute key index
        float s = sacc[t][r] + PEb[(size_t)m * N_SEQ + n];
        if (n > i0 + m) s -= 1.0e10f;
        sacc[t][r] = s * 0.0625f;
      }
    }

    // ---- partial row max over this wave's 32 columns ----
    float pm[8];
#pragma unroll
    for (int r = 0; r < 8; ++r) pm[r] = fmaxf(sacc[0][r], sacc[1][r]);
#pragma unroll
    for (int off = 1; off < 16; off <<= 1)
#pragma unroll
      for (int r = 0; r < 8; ++r)
        pm[r] = fmaxf(pm[r], __shfl_xor(pm[r], off, 32));
    if (l16 == 0)
#pragma unroll
      for (int r = 0; r < 8; ++r)
        sMax[(rw * 16 + r + 8 * half) * 2 + ch] = pm[r];
    __syncthreads();

    float alpha[8];
#pragma unroll
    for (int r = 0; r < 8; ++r) {
      int row   = rw * 16 + r + 8 * half;
      float mt  = fmaxf(sMax[row * 2 + 0], sMax[row * 2 + 1]);
      float mnw = fmaxf(m_run[r], mt);
      alpha[r]  = exp2f((m_run[r] - mnw) * 1.44269504f);
      m_run[r]  = mnw;
    }

    // ---- P = exp(sc - m): bf16 to LDS + partial row sums ----
    float psum[8];
#pragma unroll
    for (int r = 0; r < 8; ++r) psum[r] = 0.f;
#pragma unroll
    for (int t = 0; t < 2; ++t)
#pragma unroll
      for (int r = 0; r < 8; ++r) {
        float p = exp2f((sacc[t][r] - m_run[r]) * 1.44269504f);
        psum[r] += p;
        sP[(rw * 16 + r + 8 * half) * BN + ch * 32 + t * 16 + l16] = f2bf(p);
      }
#pragma unroll
    for (int off = 1; off < 16; off <<= 1)
#pragma unroll
      for (int r = 0; r < 8; ++r)
        psum[r] += __shfl_xor(psum[r], off, 32);
    if (l16 == 0)
#pragma unroll
      for (int r = 0; r < 8; ++r)
        sSum[(rw * 16 + r + 8 * half) * 2 + ch] = psum[r];

    // rescale running O and l while the barrier drains
#pragma unroll
    for (int t = 0; t < 8; ++t)
#pragma unroll
      for (int r = 0; r < 8; ++r) acc[t][r] *= alpha[r];
#pragma unroll
    for (int r = 0; r < 8; ++r) l_run[r] *= alpha[r];

    __syncthreads();
#pragma unroll
    for (int r = 0; r < 8; ++r) {
      int row = rw * 16 + r + 8 * half;
      l_run[r] += sSum[row * 2 + 0] + sSum[row * 2 + 1];
    }

    // ---- O += P @ V : 8 d-tiles, K-dim 64 = 2 steps ----
#pragma unroll
    for (int t = 0; t < 8; ++t) {
      const int d0 = ch * 128 + t * 16;
      v8f c = acc[t];
#pragma unroll
      for (int s = 0; s < 2; ++s) {
        Frag a, bb;
        const unsigned short* prow = &sP[(rw * 16 + l16) * BN + s * 32];
        a.q[0] = *(const uint4*)(prow + half * 8);
        a.q[1] = *(const uint4*)(prow + 16 + half * 8);
        const unsigned short* vrow = &sVt[(d0 + l16) * BN + s * 32];
        bb.q[0] = *(const uint4*)(vrow + half * 8);
        bb.q[1] = *(const uint4*)(vrow + 16 + half * 8);
        c = __builtin_amdgcn_wmma_f32_16x16x32_bf16(false, a.v, false, bb.v,
                                                    (short)0, c, false, false);
      }
      acc[t] = c;
    }
  }

  // ---- epilogue: normalize by l and store fp32 ----
#pragma unroll
  for (int r = 0; r < 8; ++r) {
    float inv = 1.0f / l_run[r];
    int m = rw * 16 + r + 8 * half;
#pragma unroll
    for (int t = 0; t < 8; ++t) {
      int d0 = ch * 128 + t * 16;
      Ob[(size_t)m * DK + d0 + l16] = acc[t][r] * inv;
    }
  }
}

extern "C" void kernel_launch(void* const* d_in, const int* in_sizes, int n_in,
                              void* d_out, int out_size, void* d_ws, size_t ws_size,
                              hipStream_t stream) {
  (void)in_sizes; (void)n_in; (void)out_size;
  const float* q  = (const float*)d_in[0];
  const float* k  = (const float*)d_in[1];
  const float* v  = (const float*)d_in[2];
  const float* pe = (const float*)d_in[3];
  float* out = (float*)d_out;

  dim3 grid(NTILE, B_SZ, 1);
  const size_t elems = (size_t)B_SZ * N_SEQ * DK;        // 8M per tensor
  const size_t need  = 3 * elems * sizeof(unsigned short); // 48 MB

  if (d_ws != nullptr && ws_size >= need) {
    unsigned short* qbf = (unsigned short*)d_ws;
    unsigned short* kbf = qbf + elems;
    unsigned short* vtb = kbf + elems;
    const int n4 = (int)(elems / 4);
    cvt_bf16_kernel<<<2048, NTHR, 0, stream>>>((const float4*)q, (uint2*)qbf, n4);
    cvt_bf16_kernel<<<2048, NTHR, 0, stream>>>((const float4*)k, (uint2*)kbf, n4);
    cvt_vt_kernel<<<grid, NTHR, 0, stream>>>(v, vtb);
    relattn_fa_kernel<true><<<grid, NTHR, 0, stream>>>(q, k, v, pe, out, qbf, kbf, vtb);
  } else {
    relattn_fa_kernel<false><<<grid, NTHR, 0, stream>>>(q, k, v, pe, out,
                                                        nullptr, nullptr, nullptr);
  }
}
